// LinearCRF_66314295050704
// MI455X (gfx1250) — matrix operand order repaired and checked by reading, not compiled, not run on hardware
//
#include <hip/hip_runtime.h>

// ---------------------------------------------------------------------------
// Linear-chain CRF loss on MI455X (gfx1250, wave32, WMMA).
//
// log-partition forward recursion recast as per-step GEMM:
//   p = exp(alpha - rowmax)  (f16)        [48 v_exp_f32 / step / wave]
//   S = p @ exp(Trans)                    [18 v_wmma_f32_16x16x32_f16 / step]
//   alpha' = rowmax + emit_t + log(S)     [48 v_log_f32 / step]
// 32 independent single-wave blocks, each owns 16 batch rows; exp(Trans)
// cached once in LDS as f16; intra-wave LDS transpose with s_wait_dscnt.
// Emissions/mask loads hoisted to loop top (overlap HBM latency with WMMA);
// mask applied branchlessly via fma blend; raw v_exp_f32/v_log_f32 (no
// denormal guards -- S is provably in [0.9, ~106], exp args are <= 0).
// ---------------------------------------------------------------------------

#define BB 512
#define TT 512
#define CC 96
#define TS  104   // padded LDS row stride (halves) for expT  (208B = 13*16B)
#define PST 104   // padded LDS row stride (halves) for p buffer

#define LOG2E 1.4426950408889634f
#define LN2   0.6931471805599453f

typedef _Float16 v16h __attribute__((ext_vector_type(16)));
typedef _Float16 v8h  __attribute__((ext_vector_type(8)));
typedef float    v8f  __attribute__((ext_vector_type(8)));

// Raw hardware transcendentals (backend-modeled, hazard-safe).
__device__ __forceinline__ float fast_exp(float x) {       // e^x
  return __builtin_amdgcn_exp2f(x * LOG2E);
}
__device__ __forceinline__ float fast_log(float x) {       // ln x, x normal
  return __builtin_amdgcn_logf(x) * LN2;
}

// Lane-XOR butterfly via ds_swizzle_b32 (group-of-32 mode, and=0x1f, or=0).
template <int XOR>
__device__ __forceinline__ float lane_xor_max(float v) {
  int o = __builtin_amdgcn_ds_swizzle(__float_as_int(v), (XOR << 10) | 0x1f);
  return fmaxf(v, __int_as_float(o));
}
template <int XOR>
__device__ __forceinline__ float lane_xor_add(float v) {
  int o = __builtin_amdgcn_ds_swizzle(__float_as_int(v), (XOR << 10) | 0x1f);
  return v + __int_as_float(o);
}

#define LDS_FENCE() asm volatile("s_wait_dscnt 0" ::: "memory")

__device__ __forceinline__ v16h lds_load_16h(const _Float16* p) {
  // two 16B LDS loads -> ds_load_b128 x2, concatenated into one v16h operand
  v8h lo = *(const v8h*)(p);
  v8h hi = *(const v8h*)(p + 16);
  return __builtin_shufflevector(lo, hi, 0, 1, 2, 3, 4, 5, 6, 7,
                                         8, 9, 10, 11, 12, 13, 14, 15);
}

// ---------------------------------------------------------------------------
// Kernel 1: forward algorithm -> log_den[b] for 16 batch rows per wave.
// ---------------------------------------------------------------------------
__global__ __launch_bounds__(32)
void crf_forward_kernel(const float* __restrict__ emissions,
                        const unsigned char* __restrict__ mask,
                        const float* __restrict__ start_t,
                        const float* __restrict__ end_t,
                        const float* __restrict__ trans,
                        float* __restrict__ out_logden)
{
  __shared__ __align__(16) _Float16 sT[CC * TS];   // exp(transitions), f16
  __shared__ __align__(16) _Float16 sP[16 * PST];  // p = exp(alpha - m), f16

  const int lane = (int)threadIdx.x;   // 0..31
  const int half = lane >> 4;          // C/D layout: half selects rows 0-7/8-15
  const int lc   = lane & 15;          // column within 16-wide tile
  const int b0   = (int)blockIdx.x * 16;

  // --- precompute exp(Trans) into LDS (once per wave) ---
  for (int j = lane; j < CC * CC; j += 32) {
    int r = j / CC, c = j - r * CC;
    sT[c * TS + r] = (_Float16)fast_exp(trans[j]);
  }
  LDS_FENCE();

  // --- alpha0 = start + emissions[:,0,:], kept in WMMA C/D layout ---
  // acc[w] covers columns [16w,16w+16); VGPR i holds row (i + 8*half).
  v8f acc[6];
#pragma unroll
  for (int w = 0; w < 6; ++w) {
    float st = start_t[16 * w + lc];
#pragma unroll
    for (int i = 0; i < 8; ++i) {
      int r = i + 8 * half;
      acc[w][i] = st + emissions[(size_t)(b0 + r) * TT * CC + 16 * w + lc];
    }
  }

  for (int t = 1; t < TT; ++t) {
    // ---- issue this step's global loads FIRST (overlap with WMMA phase) ----
    float ev[6][8];   // emissions in C/D layout
    float mf[8];      // mask as 0.0/1.0 per row-VGPR
#pragma unroll
    for (int i = 0; i < 8; ++i) {
      int r = i + 8 * half;
      size_t eb = ((size_t)(b0 + r) * TT + t) * CC + lc;
      mf[i] = (float)mask[(size_t)(b0 + r) * TT + t];
#pragma unroll
      for (int w = 0; w < 6; ++w) ev[w][i] = emissions[eb + 16 * w];
    }

    // ---- per-row max over 96 states ----
    float mv[8];
#pragma unroll
    for (int i = 0; i < 8; ++i) {
      float m = acc[0][i];
#pragma unroll
      for (int w = 1; w < 6; ++w) m = fmaxf(m, acc[w][i]);
      m = lane_xor_max<1>(m);
      m = lane_xor_max<2>(m);
      m = lane_xor_max<4>(m);
      m = lane_xor_max<8>(m);
      mv[i] = m;
    }

    // ---- p = exp(alpha - m) -> LDS row-major f16 (layout transpose) ----
#pragma unroll
    for (int w = 0; w < 6; ++w)
#pragma unroll
      for (int i = 0; i < 8; ++i) {
        int r = i + 8 * half;
        sP[r * PST + 16 * w + lc] = (_Float16)fast_exp(acc[w][i] - mv[i]);
      }
    LDS_FENCE();

    // ---- A operands (16x32 f16 layout): row = lc, K chunks of 32 ----
    v16h A[3];
#pragma unroll
    for (int k = 0; k < 3; ++k)
      A[k] = lds_load_16h(&sP[lc * PST + 32 * k + 8 * half]);

    // ---- S = P @ expT : 6 output column tiles, K = 3 x 32 ----
    v8f S[6];
#pragma unroll
    for (int w = 0; w < 6; ++w) {
      v8f c = {};
#pragma unroll
      for (int k = 0; k < 3; ++k) {
        // B operand (32x16 f16 layout): lane = N column, elements = K (same order as A)
        int kr = 32 * k + 8 * half;
        v16h Bv = lds_load_16h(&sT[(16 * w + lc) * TS + kr]);
        c = __builtin_amdgcn_wmma_f32_16x16x32_f16(
                false, A[k], false, Bv, (short)0, c, false, false);
      }
      S[w] = c;
    }

    // ---- alpha' = m + emit_t + log(S); branchless mask blend ----
#pragma unroll
    for (int i = 0; i < 8; ++i) {
#pragma unroll
      for (int w = 0; w < 6; ++w) {
        float na = mv[i] + ev[w][i] + fast_log(S[w][i]);
        acc[w][i] = fmaf(mf[i], na - acc[w][i], acc[w][i]);
      }
    }
  }

  // ---- log_den = logsumexp_c(alpha + end_transitions) ----
  float et[6];
#pragma unroll
  for (int w = 0; w < 6; ++w) et[w] = end_t[16 * w + lc];
#pragma unroll
  for (int i = 0; i < 8; ++i) {
    float m = acc[0][i] + et[0];
#pragma unroll
    for (int w = 1; w < 6; ++w) m = fmaxf(m, acc[w][i] + et[w]);
    m = lane_xor_max<1>(m);
    m = lane_xor_max<2>(m);
    m = lane_xor_max<4>(m);
    m = lane_xor_max<8>(m);
    float s = 0.f;
#pragma unroll
    for (int w = 0; w < 6; ++w) s += fast_exp(acc[w][i] + et[w] - m);
    s = lane_xor_add<1>(s);
    s = lane_xor_add<2>(s);
    s = lane_xor_add<4>(s);
    s = lane_xor_add<8>(s);
    if (lc == 0) out_logden[b0 + i + 8 * half] = m + fast_log(s);
  }
}

// ---------------------------------------------------------------------------
// Kernel 2: numerator (gold-path score), one block per batch row.
// ---------------------------------------------------------------------------
__global__ __launch_bounds__(128)
void crf_score_kernel(const float* __restrict__ emissions,
                      const int* __restrict__ tags,      // int (JAX x64 off)
                      const unsigned char* __restrict__ mask,
                      const float* __restrict__ start_t,
                      const float* __restrict__ end_t,
                      const float* __restrict__ trans,
                      float* __restrict__ out_num)
{
  __shared__ float rsum[128];
  __shared__ float rcnt[128];
  const int b = (int)blockIdx.x, tid = (int)threadIdx.x;

  float s = 0.f, cnt = 0.f;
  for (int t = tid; t < TT; t += 128) {
    int  tg = tags[(size_t)b * TT + t];
    float mk = (float)mask[(size_t)b * TT + t];
    cnt += mk;
    if (t == 0) {
      s += start_t[tg] + emissions[(size_t)b * TT * CC + tg];
    } else {
      int tp = tags[(size_t)b * TT + t - 1];
      s += (emissions[((size_t)b * TT + t) * CC + tg] + trans[tp * CC + tg]) * mk;
    }
  }
  rsum[tid] = s; rcnt[tid] = cnt;
  __syncthreads();
  for (int off = 64; off > 0; off >>= 1) {
    if (tid < off) { rsum[tid] += rsum[tid + off]; rcnt[tid] += rcnt[tid + off]; }
    __syncthreads();
  }
  if (tid == 0) {
    int len  = (int)(rcnt[0] + 0.5f);
    int last = tags[(size_t)b * TT + (len - 1)];
    out_num[b] = rsum[0] + end_t[last];
  }
}

// ---------------------------------------------------------------------------
// Kernel 3: mean(log_den - log_num)
// ---------------------------------------------------------------------------
__global__ __launch_bounds__(256)
void crf_final_kernel(const float* __restrict__ den,
                      const float* __restrict__ num,
                      float* __restrict__ out)
{
  __shared__ float red[256];
  const int tid = (int)threadIdx.x;
  float s = 0.f;
  for (int b = tid; b < BB; b += 256) s += den[b] - num[b];
  red[tid] = s;
  __syncthreads();
  for (int off = 128; off > 0; off >>= 1) {
    if (tid < off) red[tid] += red[tid + off];
    __syncthreads();
  }
  if (tid == 0) out[0] = red[0] / (float)BB;
}

// ---------------------------------------------------------------------------
extern "C" void kernel_launch(void* const* d_in, const int* in_sizes, int n_in,
                              void* d_out, int out_size, void* d_ws, size_t ws_size,
                              hipStream_t stream) {
  (void)in_sizes; (void)n_in; (void)out_size; (void)ws_size;
  const float*         emissions = (const float*)d_in[0];
  const int*           tags      = (const int*)d_in[1];
  const unsigned char* mask      = (const unsigned char*)d_in[2];
  const float*         start_t   = (const float*)d_in[3];
  const float*         end_t     = (const float*)d_in[4];
  const float*         trans     = (const float*)d_in[5];
  float* out = (float*)d_out;

  float* den = (float*)d_ws;        // BB floats
  float* num = den + BB;            // BB floats

  crf_forward_kernel<<<BB / 16, 32, 0, stream>>>(emissions, mask, start_t,
                                                 end_t, trans, den);
  crf_score_kernel<<<BB, 128, 0, stream>>>(emissions, tags, mask, start_t,
                                           end_t, trans, num);
  crf_final_kernel<<<1, 256, 0, stream>>>(den, num, out);
}
